// biggerbrain_26268019982564
// MI455X (gfx1250) — matrix operand order repaired and checked by reading, not compiled, not run on hardware
//
#include <hip/hip_runtime.h>

// ---------------------------------------------------------------------------
// Model dims (compile-time)
// ---------------------------------------------------------------------------
static constexpr int Bc  = 4;
static constexpr int Sc  = 512;
static constexpr int Dc  = 768;
static constexpr int Hc  = 8;
static constexpr int HDc = 96;          // D / H
static constexpr int Tc  = Bc * Sc;     // 2048 tokens
static constexpr int Vc  = 50257;
static constexpr int FFc  = 2 * Dc;     // 1536
static constexpr int FF4c = 4 * Dc;     // 3072

typedef unsigned short u16;             // bf16 storage
typedef __bf16 bf16_t;
typedef bf16_t v16bf __attribute__((ext_vector_type(16)));
typedef u16    v16u  __attribute__((ext_vector_type(16)));
typedef u16    v8u   __attribute__((ext_vector_type(8)));
typedef float  v8f   __attribute__((ext_vector_type(8)));

__device__ __forceinline__ u16 f2bf(float f) {
    unsigned int u = __float_as_uint(f);
    return (u16)((u + 0x7FFFu + ((u >> 16) & 1u)) >> 16);   // RNE
}
__device__ __forceinline__ float sigm(float x) { return 1.f / (1.f + __expf(-x)); }

// ---------------------------------------------------------------------------
// Generic batched bf16 WMMA GEMM:  C[M,N] (f32) = A[M,K] * B  (+ addC)
//   A: bf16, row-major, leading dim lda, batch offset = (z/hdiv)*sA0 + (z%hdiv)*sA1
//   B: bf16. transB=1 -> B given as [N][K] (ldb = row stride of the [N][K] view)
//            transB=0 -> B given as [K][N] (ldb = row stride)
//   Block tile 128x128, 8 waves of 32 (2x4), wave tile 64x32, K step 32.
//   Requires K % 32 == 0, A/Bt rows 16B aligned (lda,ldb multiples of 8).
// ---------------------------------------------------------------------------
__global__ __launch_bounds__(256) void k_gemm(
    const u16* __restrict__ A, int lda, long long sA0, long long sA1,
    const u16* __restrict__ Bp, int ldb, long long sB0, long long sB1, int transB,
    float* __restrict__ C, int ldc, long long sC0, long long sC1,
    const float* __restrict__ addC,
    int M, int N, int K, int hdiv)
{
    __shared__ alignas(64) u16 lA[128][32];   // [m][k]
    __shared__ alignas(64) u16 lB[128][32];   // [n][k]

    const int z  = blockIdx.z;
    const int zb = z / hdiv, zh = z - zb * hdiv;
    A  += zb * sA0 + zh * sA1;
    Bp += zb * sB0 + zh * sB1;
    C  += zb * sC0 + zh * sC1;
    if (addC) addC += zb * sC0 + zh * sC1;

    const int m0  = blockIdx.y * 128;
    const int n0  = blockIdx.x * 128;
    const int tid = threadIdx.x;
    const int lane = tid & 31;
    const int w    = tid >> 5;
    const int wm   = (w >> 2) * 64;   // wave row origin (2 rows of waves)
    const int wn   = (w & 3)  * 32;   // wave col origin (4 cols of waves)
    const int l16  = lane & 15;
    const int lh   = lane >> 4;       // lane half

    v8f acc[4][2];
    #pragma unroll
    for (int i = 0; i < 4; i++)
        #pragma unroll
        for (int j = 0; j < 2; j++)
            #pragma unroll
            for (int r = 0; r < 8; r++) acc[i][j][r] = 0.f;

    for (int k0 = 0; k0 < K; k0 += 32) {
        // ---- stage A tile: 512 chunks of 16B, 2 per thread -----------------
        #pragma unroll
        for (int cc = 0; cc < 2; cc++) {
            int chunk = tid * 2 + cc;
            int r  = chunk >> 2;
            int kk = (chunk & 3) << 3;
            int gm = m0 + r;
            uint4 val = make_uint4(0u, 0u, 0u, 0u);
            if (gm < M) {
                const u16* src = A + (long long)gm * lda + (k0 + kk);
                val = *(const uint4*)src;
                if (k0 + 32 < K) __builtin_prefetch(src + 32, 0, 1);
            }
            *(uint4*)(&lA[r][kk]) = val;
        }
        // ---- stage B tile as [n][k] ----------------------------------------
        if (transB) {
            #pragma unroll
            for (int cc = 0; cc < 2; cc++) {
                int chunk = tid * 2 + cc;
                int r  = chunk >> 2;
                int kk = (chunk & 3) << 3;
                int gn = n0 + r;
                uint4 val = make_uint4(0u, 0u, 0u, 0u);
                if (gn < N) {
                    const u16* src = Bp + (long long)gn * ldb + (k0 + kk);
                    val = *(const uint4*)src;
                    if (k0 + 32 < K) __builtin_prefetch(src + 32, 0, 1);
                }
                *(uint4*)(&lB[r][kk]) = val;
            }
        } else {
            for (int e = tid; e < 128 * 32; e += 256) {
                int n = e >> 5;
                int k = e & 31;
                int gn = n0 + n;
                lB[n][k] = (gn < N) ? Bp[(long long)(k0 + k) * ldb + gn] : (u16)0;
            }
        }
        __syncthreads();

        // ---- fragments per ISA 16-bit layouts ------------------------------
        // A 16x32: lanes 0-15 row M=l16 hold K 0-7 (v0-3) & 16-23 (v4-7);
        //          lanes 16-31 hold K 8-15 & 24-31.
        v16bf af[4], bfr[2];
        #pragma unroll
        for (int i = 0; i < 4; i++) {
            const int r = wm + i * 16 + l16;
            v8u lo = *(const v8u*)(&lA[r][lh * 8]);
            v8u hi = *(const v8u*)(&lA[r][16 + lh * 8]);
            v16u cmb = __builtin_shufflevector(lo, hi, 0,1,2,3,4,5,6,7,8,9,10,11,12,13,14,15);
            af[i] = __builtin_bit_cast(v16bf, cmb);
        }
        // B 32x16: lanes 0-15 col N=l16 hold K 0-7 & 16-23; lanes 16-31 hold K 8-15 & 24-31.
        #pragma unroll
        for (int j = 0; j < 2; j++) {
            const int n = wn + j * 16 + l16;
            v8u blo = *(const v8u*)(&lB[n][lh * 8]);
            v8u bhi = *(const v8u*)(&lB[n][16 + lh * 8]);
            v16u t = __builtin_shufflevector(blo, bhi, 0,1,2,3,4,5,6,7,8,9,10,11,12,13,14,15);
            bfr[j] = __builtin_bit_cast(v16bf, t);
        }
        #pragma unroll
        for (int i = 0; i < 4; i++)
            #pragma unroll
            for (int j = 0; j < 2; j++)
                acc[i][j] = __builtin_amdgcn_wmma_f32_16x16x32_bf16(
                    false, af[i], false, bfr[j], (short)0, acc[i][j], false, false);
        __syncthreads();
    }

    // ---- store: C/D layout: vgpr r -> row r (+8 for lane half), col = l16 --
    #pragma unroll
    for (int i = 0; i < 4; i++)
        #pragma unroll
        for (int j = 0; j < 2; j++)
            #pragma unroll
            for (int r = 0; r < 8; r++) {
                int gm = m0 + wm + i * 16 + lh * 8 + r;
                int gn = n0 + wn + j * 16 + l16;
                if (gm < M && gn < N) {
                    float vv = acc[i][j][r];
                    long long idx = (long long)gm * ldc + gn;
                    if (addC) vv += addC[idx];
                    C[idx] = vv;
                }
            }
}

// ---------------------------------------------------------------------------
// Elementwise / row-wise kernels
// ---------------------------------------------------------------------------
__global__ void k_cvt(const float* __restrict__ s, u16* __restrict__ d, long long n) {
    long long i = (long long)blockIdx.x * 256 + threadIdx.x;
    if (i < n) d[i] = f2bf(s[i]);
}
__global__ void k_cvt_t(const float* __restrict__ s, int R, int C, u16* __restrict__ d) {
    long long i = (long long)blockIdx.x * 256 + threadIdx.x;
    if (i < (long long)R * C) {
        int r = (int)(i / C), c = (int)(i - (long long)r * C);
        d[(long long)c * R + r] = f2bf(s[i]);
    }
}
__global__ void k_zero(float* __restrict__ p, long long n) {
    long long i = (long long)blockIdx.x * 256 + threadIdx.x;
    if (i < n) p[i] = 0.f;
}
__global__ void k_gather(const int* __restrict__ ids, const float* __restrict__ emb,
                         float* __restrict__ x) {
    int t = blockIdx.x;
    long long id = ids[t];
    for (int d = threadIdx.x; d < Dc; d += 256)
        x[(long long)t * Dc + d] = emb[id * Dc + d];
}
__global__ __launch_bounds__(256) void k_rmsnorm(const float* __restrict__ x,
                                                 const float* __restrict__ w,
                                                 float* __restrict__ y) {
    int t = blockIdx.x;
    const float* xr = x + (long long)t * Dc;
    float* yr = y + (long long)t * Dc;
    __shared__ float red[256];
    float s = 0.f;
    for (int d = threadIdx.x; d < Dc; d += 256) { float v = xr[d]; s += v * v; }
    red[threadIdx.x] = s; __syncthreads();
    for (int o = 128; o > 0; o >>= 1) { if (threadIdx.x < o) red[threadIdx.x] += red[threadIdx.x + o]; __syncthreads(); }
    float r = rsqrtf(red[0] / (float)Dc + 1e-6f);
    for (int d = threadIdx.x; d < Dc; d += 256) yr[d] = xr[d] * r * w[d];
}
__global__ void k_rope(float* __restrict__ x) {           // x: (B,S,H,HD) in-place
    int t = blockIdx.x;
    int s = t % Sc;
    int i = threadIdx.x;                                  // 0..383 = h*48+k
    if (i >= Hc * (HDc / 2)) return;
    int h = i / (HDc / 2);
    int k = i - h * (HDc / 2);
    float inv = __powf(10000.f, -(float)k / (float)(HDc / 2));
    float ang = (float)s * inv;
    float cs = __cosf(ang), sn = __sinf(ang);
    float* p = x + (long long)t * Dc + h * HDc;
    float x1 = p[k], x2 = p[k + HDc / 2];
    p[k]            = x1 * cs - x2 * sn;
    p[k + HDc / 2]  = x1 * sn + x2 * cs;
}
__global__ __launch_bounds__(256) void k_softmax(const float* __restrict__ sc,
                                                 u16* __restrict__ pb) {
    long long row = blockIdx.x;                 // (b*H+h)*S + i
    int i = (int)(row % Sc);
    const float* srow = sc + row * Sc;
    u16* prow = pb + row * Sc;
    const float scale = 0.1020620726159658f;    // 1/sqrt(96)
    __shared__ float red[256];
    int tid = threadIdx.x;
    float mx = -3.4e38f;
    for (int j = tid; j <= i; j += 256) mx = fmaxf(mx, srow[j] * scale);
    red[tid] = mx; __syncthreads();
    for (int o = 128; o > 0; o >>= 1) { if (tid < o) red[tid] = fmaxf(red[tid], red[tid + o]); __syncthreads(); }
    mx = red[0]; __syncthreads();
    float sum = 0.f;
    for (int j = tid; j <= i; j += 256) sum += __expf(srow[j] * scale - mx);
    red[tid] = sum; __syncthreads();
    for (int o = 128; o > 0; o >>= 1) { if (tid < o) red[tid] += red[tid + o]; __syncthreads(); }
    float inv = 1.f / red[0];
    for (int j = tid; j < Sc; j += 256) {
        float p = (j <= i) ? __expf(srow[j] * scale - mx) * inv : 0.f;
        prow[j] = f2bf(p);
    }
}
__global__ void k_silu_mul(const float* __restrict__ in, u16* __restrict__ out,
                           int ff, long long n) {
    long long i = (long long)blockIdx.x * 256 + threadIdx.x;
    if (i < n) {
        long long t = i / ff; int f = (int)(i - t * ff);
        float a  = in[t * (2LL * ff) + f];
        float b2 = in[t * (2LL * ff) + ff + f];
        out[i] = f2bf(a * sigm(a) * b2);
    }
}
__global__ void k_last(const float* __restrict__ y, float* __restrict__ last) {
    int b = blockIdx.x;
    for (int d = threadIdx.x; d < Dc; d += 256)
        last[(long long)b * Dc + d] = y[(((long long)b * Sc) + (Sc - 1)) * Dc + d];
}
__global__ void k_bcast(const float* __restrict__ mem, float* __restrict__ out) {
    int t = blockIdx.x; int b = t / Sc;
    for (int d = threadIdx.x; d < Dc; d += 256)
        out[(long long)t * Dc + d] = mem[(long long)b * Dc + d];
}
__global__ __launch_bounds__(256) void k_gru_cell(const float* __restrict__ x,
    const float* __restrict__ hprev, const float* __restrict__ wih,
    const float* __restrict__ whh, const float* __restrict__ bih,
    const float* __restrict__ bhh, float* __restrict__ hout) {
    int b = blockIdx.x;
    __shared__ float xs[Dc], hs[Dc];
    __shared__ float gi[3 * Dc], gh[3 * Dc];
    int tid = threadIdx.x;
    for (int d = tid; d < Dc; d += 256) {
        xs[d] = x[(long long)b * Dc + d];
        hs[d] = hprev[(long long)b * Dc + d];
    }
    __syncthreads();
    for (int j = tid; j < 3 * Dc; j += 256) {
        const float* wi = wih + (long long)j * Dc;
        const float* wh = whh + (long long)j * Dc;
        float si = bih[j], sh = bhh[j];
        for (int d = 0; d < Dc; d++) { si += wi[d] * xs[d]; sh += wh[d] * hs[d]; }
        gi[j] = si; gh[j] = sh;
    }
    __syncthreads();
    for (int d = tid; d < Dc; d += 256) {
        float r  = sigm(gi[d] + gh[d]);
        float zz = sigm(gi[Dc + d] + gh[Dc + d]);
        float n  = tanhf(gi[2 * Dc + d] + r * gh[2 * Dc + d]);
        hout[(long long)b * Dc + d] = (1.f - zz) * n + zz * hs[d];
    }
}
__global__ __launch_bounds__(1024) void k_gru_seq(const float* __restrict__ gi,
    const float* __restrict__ whh, const float* __restrict__ bih,
    const float* __restrict__ bhh, float* __restrict__ out) {
    int b = blockIdx.x;
    __shared__ float h[Dc];
    __shared__ float gh[3 * Dc];
    __shared__ float hn[Dc];
    int tid = threadIdx.x;
    for (int d = tid; d < Dc; d += 1024) h[d] = 0.f;
    __syncthreads();
    for (int t = 0; t < Sc; t++) {
        const float* gir = gi + ((long long)b * Sc + t) * (3 * Dc);
        for (int j = tid; j < 3 * Dc; j += 1024) {
            const float* wr = whh + (long long)j * Dc;
            float s = bhh[j];
            for (int d = 0; d < Dc; d++) s += wr[d] * h[d];
            gh[j] = s;
        }
        __syncthreads();
        for (int d = tid; d < Dc; d += 1024) {
            float r  = sigm(gir[d] + bih[d] + gh[d]);
            float zz = sigm(gir[Dc + d] + bih[Dc + d] + gh[Dc + d]);
            float n  = tanhf(gir[2 * Dc + d] + bih[2 * Dc + d] + r * gh[2 * Dc + d]);
            hn[d] = (1.f - zz) * n + zz * h[d];
        }
        __syncthreads();
        float* orow = out + ((long long)b * Sc + t) * Dc;
        for (int d = tid; d < Dc; d += 1024) { h[d] = hn[d]; orow[d] = hn[d]; }
        __syncthreads();
    }
}
__global__ __launch_bounds__(256) void k_memgate(const float* __restrict__ last,
    const float* __restrict__ sgw, const float* __restrict__ sgb,
    const float* __restrict__ hn, float* __restrict__ mem) {
    int b = blockIdx.x;
    __shared__ float red[256];
    float s = 0.f;
    for (int d = threadIdx.x; d < Dc; d += 256) s += last[(long long)b * Dc + d] * sgw[d];
    red[threadIdx.x] = s; __syncthreads();
    for (int o = 128; o > 0; o >>= 1) { if (threadIdx.x < o) red[threadIdx.x] += red[threadIdx.x + o]; __syncthreads(); }
    float wg = sigm(red[0] + sgb[0]);
    for (int d = threadIdx.x; d < Dc; d += 256) {
        long long i = (long long)b * Dc + d;
        mem[i] = wg * hn[i] + (1.f - wg) * mem[i];
    }
}
__global__ __launch_bounds__(256) void k_mol_gate(const float* __restrict__ x,
    const float* __restrict__ gw, float* __restrict__ g) {
    int t = blockIdx.x;
    __shared__ float r0[256], r1[256];
    float a = 0.f, b2 = 0.f;
    for (int d = threadIdx.x; d < Dc; d += 256) {
        float v = x[(long long)t * Dc + d];
        a  += v * gw[d * 2];
        b2 += v * gw[d * 2 + 1];
    }
    r0[threadIdx.x] = a; r1[threadIdx.x] = b2; __syncthreads();
    for (int o = 128; o > 0; o >>= 1) {
        if (threadIdx.x < o) { r0[threadIdx.x] += r0[threadIdx.x + o]; r1[threadIdx.x] += r1[threadIdx.x + o]; }
        __syncthreads();
    }
    if (threadIdx.x == 0) {
        float g0 = r0[0], g1 = r1[0];
        float m = fmaxf(g0, g1);
        float e0 = __expf(g0 - m), e1 = __expf(g1 - m);
        float inv = 1.f / (e0 + e1);
        g[t * 2] = e0 * inv; g[t * 2 + 1] = e1 * inv;
    }
}
__global__ void k_mol_combine(const float* __restrict__ resid, const float* __restrict__ g,
    const float* __restrict__ e0, const float* __restrict__ e1,
    float* __restrict__ out, long long n) {
    long long i = (long long)blockIdx.x * 256 + threadIdx.x;
    if (i < n) {
        long long t = i / Dc;
        out[i] = resid[i] + g[t * 2] * e0[i] + g[t * 2 + 1] * e1[i];
    }
}
__global__ __launch_bounds__(256) void k_final_gate(const float* __restrict__ anch,
    const float* __restrict__ y, const float* __restrict__ w,
    const float* __restrict__ bptr, float* __restrict__ out) {
    int t = blockIdx.x;
    __shared__ float red[256];
    float s = 0.f;
    for (int d = threadIdx.x; d < Dc; d += 256)
        s += anch[(long long)t * Dc + d] * w[d] + y[(long long)t * Dc + d] * w[Dc + d];
    red[threadIdx.x] = s; __syncthreads();
    for (int o = 128; o > 0; o >>= 1) { if (threadIdx.x < o) red[threadIdx.x] += red[threadIdx.x + o]; __syncthreads(); }
    float g = sigm(red[0] + bptr[0]);
    for (int d = threadIdx.x; d < Dc; d += 256) {
        long long i = (long long)t * Dc + d;
        out[i] = g * y[i] + (1.f - g) * anch[i];
    }
}

// ---------------------------------------------------------------------------
// Host orchestration
// ---------------------------------------------------------------------------
enum {
    IN_IDS = 0, IN_EMBED = 1,
    IN_A1 = 2, IN_A2 = 6, IN_A3 = 10, IN_A4 = 14, IN_PRE = 18, IN_MA1 = 22, IN_MA2 = 26,
    IN_FFNPRE_W1 = 30, IN_FFNPRE_W2 = 31,
    IN_MI2_W1 = 32, IN_MI2_W2 = 33, IN_MI4_W1 = 34, IN_MI4_W2 = 35,
    IN_MOL1_GATE = 36, IN_MOL1_E0W1 = 37, IN_MOL1_E0W2 = 38, IN_MOL1_E1W1 = 39, IN_MOL1_E1W2 = 40,
    IN_MOL3_GATE = 41, IN_MOL3_E0W1 = 42, IN_MOL3_E0W2 = 43, IN_MOL3_E1W1 = 44, IN_MOL3_E1W2 = 45,
    IN_MEM_WIH = 46, IN_MEM_WHH = 47, IN_MEM_BIH = 48, IN_MEM_BHH = 49,
    IN_GRU_WIH = 50, IN_GRU_WHH = 51, IN_GRU_BIH = 52, IN_GRU_BHH = 53,
    IN_SG_W = 54, IN_SG_B = 55,
    IN_NORMPRE = 56, IN_NORM0 = 57, IN_NORM1 = 58, IN_NORM2 = 59, IN_NORM3 = 60, IN_NORM4 = 61,
    IN_MM1_W = 62, IN_MM1_B = 63
};

struct Ctx {
    hipStream_t st;
    const float* inF[64];
    const int*   ids;
    u16* wA[7][4];                 // A1,A2,A3,A4,PRE,MA1,MA2 x {q,k,v,o} transposed bf16
    u16 *wPre1, *wPre2, *wMi21, *wMi22, *wMi41, *wMi42;
    u16 *wM1[4], *wM3[4];          // e0w1t,e0w2t,e1w1t,e1w2t
    u16 *gruWih, *embedBf;
    float *X, *A, *Bb, *Cc, *ANCH, *SEQ, *M1, *Qf, *Kf, *Vf, *ATT, *BIG, *GI, *SC;
    float *E0, *E1, *G2, *LAST, *HN, *MEM;
    u16 *BFI, *QB, *KB, *VB, *PB;
};

static void gemm(hipStream_t st, const u16* A, int lda, long long sA0, long long sA1,
                 const u16* Bm, int ldb, long long sB0, long long sB1, int tB,
                 float* C, int ldc, long long sC0, long long sC1, const float* add,
                 int M, int N, int K, int batch, int hdiv) {
    dim3 g((unsigned)((N + 127) / 128), (unsigned)((M + 127) / 128), (unsigned)batch);
    k_gemm<<<g, 256, 0, st>>>(A, lda, sA0, sA1, Bm, ldb, sB0, sB1, tB,
                              C, ldc, sC0, sC1, add, M, N, K, hdiv);
}
static void cvt(hipStream_t st, const float* s, u16* d, long long n) {
    k_cvt<<<(unsigned)((n + 255) / 256), 256, 0, st>>>(s, d, n);
}
static void cvtT(hipStream_t st, const float* s, int R, int C, u16* d) {
    long long n = (long long)R * C;
    k_cvt_t<<<(unsigned)((n + 255) / 256), 256, 0, st>>>(s, R, C, d);
}

// attention(xq, xkv) with optional RoPE and residual, -> out (Tc x Dc)
static void attention(Ctx& c, const float* xq, const float* xkv, u16* const wset[4],
                      bool use_rope, const float* resid, float* out) {
    cvt(c.st, xq, c.BFI, (long long)Tc * Dc);
    gemm(c.st, c.BFI, Dc, 0, 0, wset[0], Dc, 0, 0, 1, c.Qf, Dc, 0, 0, nullptr, Tc, Dc, Dc, 1, 1);
    if (xkv != xq) cvt(c.st, xkv, c.BFI, (long long)Tc * Dc);
    gemm(c.st, c.BFI, Dc, 0, 0, wset[1], Dc, 0, 0, 1, c.Kf, Dc, 0, 0, nullptr, Tc, Dc, Dc, 1, 1);
    gemm(c.st, c.BFI, Dc, 0, 0, wset[2], Dc, 0, 0, 1, c.Vf, Dc, 0, 0, nullptr, Tc, Dc, Dc, 1, 1);
    if (use_rope) {
        k_rope<<<Tc, 384, 0, c.st>>>(c.Qf);
        k_rope<<<Tc, 384, 0, c.st>>>(c.Kf);
    }
    cvt(c.st, c.Qf, c.QB, (long long)Tc * Dc);
    cvt(c.st, c.Kf, c.KB, (long long)Tc * Dc);
    cvt(c.st, c.Vf, c.VB, (long long)Tc * Dc);
    // scores[b,h,q,k] = q . k   (batched over b*H, per-head offsets)
    gemm(c.st, c.QB, Dc, (long long)Sc * Dc, HDc,
              c.KB, Dc, (long long)Sc * Dc, HDc, 1,
              c.SC, Sc, (long long)Hc * Sc * Sc, (long long)Sc * Sc, nullptr,
              Sc, Sc, HDc, Bc * Hc, Hc);
    k_softmax<<<Bc * Hc * Sc, 256, 0, c.st>>>(c.SC, c.PB);
    // out[b,q,h,hd] = probs . v   (B as [K][N] with ldb = Dc)
    gemm(c.st, c.PB, Sc, (long long)Hc * Sc * Sc, (long long)Sc * Sc,
              c.VB, Dc, (long long)Sc * Dc, HDc, 0,
              c.ATT, Dc, (long long)Sc * Dc, HDc, nullptr,
              Sc, HDc, Sc, Bc * Hc, Hc);
    cvt(c.st, c.ATT, c.BFI, (long long)Tc * Dc);
    gemm(c.st, c.BFI, Dc, 0, 0, wset[3], Dc, 0, 0, 1, out, Dc, 0, 0, resid, Tc, Dc, Dc, 1, 1);
}

static void swiglu(Ctx& c, const float* x, const u16* w1t, const u16* w2t, int ff,
                   const float* resid, float* out) {
    cvt(c.st, x, c.BFI, (long long)Tc * Dc);
    gemm(c.st, c.BFI, Dc, 0, 0, w1t, Dc, 0, 0, 1, c.BIG, 2 * ff, 0, 0, nullptr, Tc, 2 * ff, Dc, 1, 1);
    long long n = (long long)Tc * ff;
    k_silu_mul<<<(unsigned)((n + 255) / 256), 256, 0, c.st>>>(c.BIG, c.BFI, ff, n);
    gemm(c.st, c.BFI, ff, 0, 0, w2t, ff, 0, 0, 1, out, Dc, 0, 0, resid, Tc, Dc, ff, 1, 1);
}

static void mol(Ctx& c, const float* x, const float* gateW, u16* const we[4],
                const float* resid, float* out) {
    k_mol_gate<<<Tc, 256, 0, c.st>>>(x, gateW, c.G2);
    swiglu(c, x, we[0], we[1], FF4c, nullptr, c.E0);
    swiglu(c, x, we[2], we[3], FF4c, nullptr, c.E1);
    long long n = (long long)Tc * Dc;
    k_mol_combine<<<(unsigned)((n + 255) / 256), 256, 0, c.st>>>(resid, c.G2, c.E0, c.E1, out, n);
}

extern "C" void kernel_launch(void* const* d_in, const int* in_sizes, int n_in,
                              void* d_out, int out_size, void* d_ws, size_t ws_size,
                              hipStream_t stream) {
    (void)in_sizes; (void)n_in; (void)out_size; (void)ws_size;
    Ctx c{};
    c.st = stream;
    c.ids = (const int*)d_in[0];
    for (int i = 1; i < 64; i++) c.inF[i] = (const float*)d_in[i];

    // ---- workspace bump allocator ----
    char* wsb = (char*)d_ws;
    size_t off = 0;
    auto alloc = [&](size_t bytes) -> void* {
        void* p = wsb + off;
        off = (off + bytes + 255) & ~(size_t)255;
        return p;
    };

    // ---- convert weights to bf16 (transposed to [N][K] where needed) ----
    const int setBase[7] = {IN_A1, IN_A2, IN_A3, IN_A4, IN_PRE, IN_MA1, IN_MA2};
    for (int s = 0; s < 7; s++)
        for (int m = 0; m < 4; m++) {
            c.wA[s][m] = (u16*)alloc((size_t)Dc * Dc * 2);
            cvtT(stream, c.inF[setBase[s] + m], Dc, Dc, c.wA[s][m]);
        }
    c.wPre1 = (u16*)alloc((size_t)2 * FFc * Dc * 2);  cvtT(stream, c.inF[IN_FFNPRE_W1], Dc, 2 * FFc, c.wPre1);
    c.wPre2 = (u16*)alloc((size_t)Dc * FFc * 2);      cvtT(stream, c.inF[IN_FFNPRE_W2], FFc, Dc, c.wPre2);
    c.wMi21 = (u16*)alloc((size_t)2 * FF4c * Dc * 2); cvtT(stream, c.inF[IN_MI2_W1], Dc, 2 * FF4c, c.wMi21);
    c.wMi22 = (u16*)alloc((size_t)Dc * FF4c * 2);     cvtT(stream, c.inF[IN_MI2_W2], FF4c, Dc, c.wMi22);
    c.wMi41 = (u16*)alloc((size_t)2 * FF4c * Dc * 2); cvtT(stream, c.inF[IN_MI4_W1], Dc, 2 * FF4c, c.wMi41);
    c.wMi42 = (u16*)alloc((size_t)Dc * FF4c * 2);     cvtT(stream, c.inF[IN_MI4_W2], FF4c, Dc, c.wMi42);
    const int molIdx[2][4] = {{IN_MOL1_E0W1, IN_MOL1_E0W2, IN_MOL1_E1W1, IN_MOL1_E1W2},
                              {IN_MOL3_E0W1, IN_MOL3_E0W2, IN_MOL3_E1W1, IN_MOL3_E1W2}};
    for (int mset = 0; mset < 2; mset++)
        for (int e = 0; e < 2; e++) {
            u16* w1t = (u16*)alloc((size_t)2 * FF4c * Dc * 2);
            u16* w2t = (u16*)alloc((size_t)Dc * FF4c * 2);
            cvtT(stream, c.inF[molIdx[mset][e * 2 + 0]], Dc, 2 * FF4c, w1t);
            cvtT(stream, c.inF[molIdx[mset][e * 2 + 1]], FF4c, Dc, w2t);
            if (mset == 0) { c.wM1[e * 2] = w1t; c.wM1[e * 2 + 1] = w2t; }
            else           { c.wM3[e * 2] = w1t; c.wM3[e * 2 + 1] = w2t; }
        }
    c.gruWih  = (u16*)alloc((size_t)3 * Dc * Dc * 2);
    cvt(stream, c.inF[IN_GRU_WIH], c.gruWih, (long long)3 * Dc * Dc);   // [3D][D] == [N][K]
    c.embedBf = (u16*)alloc((size_t)Vc * Dc * 2);
    cvt(stream, c.inF[IN_EMBED], c.embedBf, (long long)Vc * Dc);        // [V][D] == [N][K]

    // ---- activation buffers ----
    const size_t TD = (size_t)Tc * Dc * 4;
    c.X    = (float*)alloc(TD);  c.A   = (float*)alloc(TD);  c.Bb  = (float*)alloc(TD);
    c.Cc   = (float*)alloc(TD);  c.ANCH = (float*)alloc(TD); c.SEQ = (float*)alloc(TD);
    c.M1   = (float*)alloc(TD);  c.Qf  = (float*)alloc(TD);  c.Kf  = (float*)alloc(TD);
    c.Vf   = (float*)alloc(TD);  c.ATT = (float*)alloc(TD);
    c.BIG  = (float*)alloc((size_t)Tc * 2 * FF4c * 4);
    c.GI   = (float*)alloc((size_t)Tc * 3 * Dc * 4);
    c.SC   = (float*)alloc((size_t)Bc * Hc * Sc * Sc * 4);
    c.E0   = (float*)alloc(TD);  c.E1  = (float*)alloc(TD);
    c.G2   = (float*)alloc((size_t)Tc * 2 * 4);
    c.LAST = (float*)alloc((size_t)Bc * Dc * 4);
    c.HN   = (float*)alloc((size_t)Bc * Dc * 4);
    c.MEM  = (float*)alloc((size_t)Bc * Dc * 4);
    c.BFI  = (u16*)alloc((size_t)Tc * 2 * FF4c * 2);
    c.QB   = (u16*)alloc((size_t)Tc * Dc * 2);
    c.KB   = (u16*)alloc((size_t)Tc * Dc * 2);
    c.VB   = (u16*)alloc((size_t)Tc * Dc * 2);
    c.PB   = (u16*)alloc((size_t)Bc * Hc * Sc * Sc * 2);

    // ================= forward pass =================
    k_gather<<<Tc, 256, 0, stream>>>(c.ids, c.inF[IN_EMBED], c.X);
    k_zero<<<(unsigned)((Bc * Dc + 255) / 256), 256, 0, stream>>>(c.MEM, (long long)Bc * Dc);

    // y = attn(x, x, attA1, rope)
    attention(c, c.X, c.X, c.wA[0], true, nullptr, c.Bb);
    // memory write 1
    k_last<<<Bc, 256, 0, stream>>>(c.Bb, c.LAST);
    k_gru_cell<<<Bc, 256, 0, stream>>>(c.LAST, c.MEM, c.inF[IN_MEM_WIH], c.inF[IN_MEM_WHH],
                                       c.inF[IN_MEM_BIH], c.inF[IN_MEM_BHH], c.HN);
    k_memgate<<<Bc, 256, 0, stream>>>(c.LAST, c.inF[IN_SG_W], c.inF[IN_SG_B], c.HN, c.MEM);
    k_bcast<<<Tc, 256, 0, stream>>>(c.MEM, c.M1);

    // y = rmsnorm(y, normPre); y = attn(y,y,attPre,rope)+y; y = y + swiglu(y, ffnPre)
    k_rmsnorm<<<Tc, 256, 0, stream>>>(c.Bb, c.inF[IN_NORMPRE], c.A);
    attention(c, c.A, c.A, c.wA[4], true, c.A, c.Cc);
    swiglu(c, c.Cc, c.wPre1, c.wPre2, FFc, c.Cc, c.ANCH);          // anchor = residual

    // seq_context = gru_seq(y)
    cvt(stream, c.ANCH, c.BFI, (long long)Tc * Dc);
    gemm(stream, c.BFI, Dc, 0, 0, c.gruWih, Dc, 0, 0, 1, c.GI, 3 * Dc, 0, 0, nullptr,
         Tc, 3 * Dc, Dc, 1, 1);
    k_gru_seq<<<Bc, 1024, 0, stream>>>(c.GI, c.inF[IN_GRU_WHH], c.inF[IN_GRU_BIH],
                                       c.inF[IN_GRU_BHH], c.SEQ);

    // y = attn(rmsnorm(y,norm0), ., attA1, rope) + residual
    k_rmsnorm<<<Tc, 256, 0, stream>>>(c.ANCH, c.inF[IN_NORM0], c.A);
    attention(c, c.A, c.A, c.wA[0], true, c.ANCH, c.Bb);
    // y = attn(y, m1_out, MA1)    (no residual)
    attention(c, c.Bb, c.M1, c.wA[5], false, nullptr, c.A);
    // y = attn(y, seq_context, MA2)
    attention(c, c.A, c.SEQ, c.wA[6], false, nullptr, c.Bb);
    // (reference's second mem-cell update is dead code: mem1 never read again)

    // y = rmsnorm(y,norm1); y = y + mol(y, mol1)
    k_rmsnorm<<<Tc, 256, 0, stream>>>(c.Bb, c.inF[IN_NORM1], c.A);
    mol(c, c.A, c.inF[IN_MOL1_GATE], c.wM1, c.A, c.Bb);
    // y = rmsnorm(y,norm2); y = attn(y,y,attA2,rope)+y; y = y + swiglu(y, ffnMi2)
    k_rmsnorm<<<Tc, 256, 0, stream>>>(c.Bb, c.inF[IN_NORM2], c.A);
    attention(c, c.A, c.A, c.wA[1], true, c.A, c.Bb);
    swiglu(c, c.Bb, c.wMi21, c.wMi22, FF4c, c.Bb, c.A);
    // y = rmsnorm(y,norm3); y = attn(y,y,attA3,rope)+y; y = y + mol(y, mol3)
    k_rmsnorm<<<Tc, 256, 0, stream>>>(c.A, c.inF[IN_NORM3], c.Bb);
    attention(c, c.Bb, c.Bb, c.wA[2], true, c.Bb, c.A);
    mol(c, c.A, c.inF[IN_MOL3_GATE], c.wM3, c.A, c.Bb);
    // y = rmsnorm(y,norm4); y = attn(y,y,attA4,rope)+y; y = y + swiglu(y, ffnMi4)
    k_rmsnorm<<<Tc, 256, 0, stream>>>(c.Bb, c.inF[IN_NORM4], c.A);
    attention(c, c.A, c.A, c.wA[3], true, c.A, c.Bb);
    swiglu(c, c.Bb, c.wMi41, c.wMi42, FF4c, c.Bb, c.A);

    // final gate mix with anchor, then logits = y @ embed^T
    k_final_gate<<<Tc, 256, 0, stream>>>(c.ANCH, c.A, c.inF[IN_MM1_W], c.inF[IN_MM1_B], c.Cc);
    cvt(stream, c.Cc, c.BFI, (long long)Tc * Dc);
    gemm(stream, c.BFI, Dc, 0, 0, c.embedBf, Dc, 0, 0, 1,
         (float*)d_out, Vc, 0, 0, nullptr, Tc, Vc, Dc, 1, 1);
}
